// CustomBidirectionalLSTM_76063870812454
// MI455X (gfx1250) — matrix operand order repaired and checked by reading, not compiled, not run on hardware
//
#include <hip/hip_runtime.h>
#include <math.h>

// ---------------------------------------------------------------------------
// Bidirectional 2-layer LSTM for MI455X (gfx1250), wave32 + WMMA bf16.
//   B=32, T=512, IN=512, H=512, 4H=2048.
// Strategy:
//   1) Pack X and all weights to bf16 (weights into WMMA B-fragment layout).
//   2) gx = X @ W_ih^T + (b_ih+b_hh) as a big parallel WMMA GEMM (per dir).
//   3) Recurrence: one persistent 256-thread workgroup per direction;
//      h double-buffered in LDS (bf16), c in VGPRs, W_hh streamed from L2,
//      v_wmma_f32_16x16x32_bf16 for h @ W_hh^T, s_barrier per timestep.
//   All four B-fragments are loaded into distinct values before the four
//   WMMAs so loads clause together and waits decrement instead of stalling.
// ---------------------------------------------------------------------------

typedef __attribute__((ext_vector_type(16))) __bf16 v16bf;
typedef __attribute__((ext_vector_type(8)))  __bf16 v8bf;
typedef __attribute__((ext_vector_type(8)))  float  v8f;

#define DEVINL static __device__ __forceinline__

DEVINL __bf16 f2bf(float f) {
  unsigned u = __builtin_bit_cast(unsigned, f);
  unsigned r = (u + 0x7FFFu + ((u >> 16) & 1u)) >> 16;   // RNE
  unsigned short s = (unsigned short)r;
  return __builtin_bit_cast(__bf16, s);
}
DEVINL float bf2f(__bf16 b) {
  unsigned short s = __builtin_bit_cast(unsigned short, b);
  unsigned u = ((unsigned)s) << 16;
  return __builtin_bit_cast(float, u);
}
DEVINL v16bf mk_frag(v8bf lo, v8bf hi) {
  v16bf r;
#pragma unroll
  for (int i = 0; i < 8; ++i) { r[i] = lo[i]; r[i + 8] = hi[i]; }
  return r;
}
DEVINL float sigmoidf_(float x) { return 1.0f / (1.0f + __expf(-x)); }
DEVINL float tanh_fast(float x) {
  float e = __expf(-2.0f * fabsf(x));
  float t = (1.0f - e) / (1.0f + e);
  return x < 0.0f ? -t : t;
}

// ---------------- pack X: [B=32][T=512][512] f32 -> [T*32+b][512] bf16 ------
__global__ __launch_bounds__(256) void pack_x_kernel(const float* __restrict__ X,
                                                     __bf16* __restrict__ Xbf) {
  const int idx = blockIdx.x * 256 + threadIdx.x;   // total = 32*512*512
  if (idx >= 32 * 512 * 512) return;
  const int d = idx & 511;
  const int t = (idx >> 9) & 511;
  const int b = idx >> 18;
  Xbf[((size_t)t * 32 + b) * 512 + d] = f2bf(X[idx]);
}

// ---------------- pack W [2048][K] f32 -> WMMA B-fragment tiles (bf16) ------
// Layout: dst[((kstep*128 + nt)*32 + lane)*16 + i], lane: n = nt*16+(lane&15),
//         k = kstep*32 + (lane>>4)*16 + i  (B is 32x16 K-major per lane).
__global__ __launch_bounds__(256) void pack_w_kernel(const float* __restrict__ W,
                                                     __bf16* __restrict__ dst, int K) {
  const int idx = blockIdx.x * 256 + threadIdx.x;
  if (idx >= 2048 * K) return;
  const int i    = idx & 15;
  const int lane = (idx >> 4) & 31;
  const int nt   = (idx >> 9) & 127;
  const int ks   = idx >> 16;
  const int n = nt * 16 + (lane & 15);
  const int k = ks * 32 + ((lane >> 4) << 4) + i;
  dst[idx] = f2bf(W[(size_t)n * K + k]);
}

// ---------------- combined bias ---------------------------------------------
__global__ __launch_bounds__(256) void bias_kernel(const float* __restrict__ a,
                                                   const float* __restrict__ b,
                                                   float* __restrict__ o) {
  const int i = blockIdx.x * 256 + threadIdx.x;
  if (i < 2048) o[i] = a[i] + b[i];
}

// ---------------- gx = A[M][K] @ Wp + bias  ->  C[M][2048] bf16 -------------
// One wave computes a 16x64 strip (4 N-tiles share one A-fragment).
__global__ __launch_bounds__(256) void gemm_gx_kernel(
    const __bf16* __restrict__ A, const __bf16* __restrict__ Wp,
    const float* __restrict__ bias, __bf16* __restrict__ C, int M, int K) {
  const int lane = threadIdx.x & 31;
  const int gw   = blockIdx.x * 8 + (threadIdx.x >> 5);
  const int mt = gw >> 5;           // M tile
  const int ng = gw & 31;           // group of 4 N tiles
  if (mt * 16 >= M) return;
  const int ksteps = K >> 5;
  const int mrow = mt * 16 + (lane & 15);
  const int kb   = (lane >> 4) << 4;
  v8f acc[4];
#pragma unroll
  for (int j = 0; j < 4; ++j)
#pragma unroll
    for (int r = 0; r < 8; ++r) acc[j][r] = 0.0f;

  for (int ks = 0; ks < ksteps; ++ks) {
    const __bf16* ap = A + (size_t)mrow * K + ks * 32 + kb;
    v16bf a = mk_frag(*(const v8bf*)ap, *(const v8bf*)(ap + 8));
    v16bf b[4];
#pragma unroll
    for (int j = 0; j < 4; ++j) {              // all loads first -> one clause
      const __bf16* bp = Wp + (((size_t)ks * 128 + (ng * 4 + j)) * 32 + lane) * 16;
      b[j] = mk_frag(*(const v8bf*)bp, *(const v8bf*)(bp + 8));
    }
#pragma unroll
    for (int j = 0; j < 4; ++j)
      acc[j] = __builtin_amdgcn_wmma_f32_16x16x32_bf16(
          false, a, false, b[j], (short)0, acc[j], false, false);
  }
  const int rbase = mt * 16 + ((lane >> 4) << 3);
  const int nc0 = lane & 15;
#pragma unroll
  for (int j = 0; j < 4; ++j) {
    const int ncol = (ng * 4 + j) * 16 + nc0;
    const float bv = bias[ncol];
#pragma unroll
    for (int r = 0; r < 8; ++r)
      C[(size_t)(rbase + r) * 2048 + ncol] = f2bf(acc[j][r] + bv);
  }
}

// ---------------- persistent recurrence: one workgroup per direction --------
__global__ __launch_bounds__(256) void lstm_layer_kernel(
    const __bf16* __restrict__ gx_f, const __bf16* __restrict__ gx_r,
    const __bf16* __restrict__ Wp_f, const __bf16* __restrict__ Wp_r,
    __bf16* __restrict__ hs_bf,    // layer0: [T][32][1024] (dir offset 512); else null
    float*  __restrict__ out_f32,  // layer1: final_output [32][512][1024]; else null
    float*  __restrict__ hT_base, float* __restrict__ cT_base) {
  const int dir = blockIdx.x;
  const __bf16* gx = dir ? gx_r : gx_f;
  const __bf16* Wp = dir ? Wp_r : Wp_f;

  __shared__ __bf16 hbuf[2][32 * 512];          // 64 KB, double-buffered h (bf16)
  for (int i = threadIdx.x; i < 32 * 512; i += 256)
    hbuf[0][i] = __builtin_bit_cast(__bf16, (unsigned short)0);
  __syncthreads();

  const int wave = threadIdx.x >> 5;
  const int lane = threadIdx.x & 31;
  const int nc0 = lane & 15;
  const int kb  = (lane >> 4) << 4;

  float cst[8][8];                               // c-state, 8 tile-groups/wave
#pragma unroll
  for (int g = 0; g < 8; ++g)
#pragma unroll
    for (int r = 0; r < 8; ++r) cst[g][r] = 0.0f;

  for (int s = 0; s < 512; ++s) {
    const int t = dir ? (511 - s) : s;
    const __bf16* hread  = hbuf[s & 1];
    __bf16*       hwrite = hbuf[(s & 1) ^ 1];
    const __bf16* gxt = gx + (size_t)t * 32 * 2048;

#pragma unroll
    for (int g = 0; g < 8; ++g) {
      const int gi = wave * 8 + g;               // 64 groups total
      const int mt = gi & 1;                     // batch tile (0..1)
      const int jt = gi >> 1;                    // hidden tile (0..31)
      const int rbase = mt * 16 + ((lane >> 4) << 3);
      const int mrow  = mt * 16 + nc0;

      v8f acc[4];                                // i,f,g,o tiles
#pragma unroll
      for (int q = 0; q < 4; ++q) {
        const int ncol = q * 512 + jt * 16 + nc0;
#pragma unroll
        for (int r = 0; r < 8; ++r)
          acc[q][r] = bf2f(gxt[(size_t)(rbase + r) * 2048 + ncol]);
      }

      for (int ks = 0; ks < 16; ++ks) {
        const __bf16* hp = hread + mrow * 512 + ks * 32 + kb;
        v16bf a = mk_frag(*(const v8bf*)hp, *(const v8bf*)(hp + 8));
        if (ks < 15)                             // warm WGP$ for next k-slab
          __builtin_prefetch(Wp + (((size_t)(ks + 1) * 128 + jt) * 32 + lane) * 16, 0, 1);
        v16bf b[4];
#pragma unroll
        for (int q = 0; q < 4; ++q) {            // all loads first -> one clause
          const __bf16* bp = Wp + (((size_t)ks * 128 + (q * 32 + jt)) * 32 + lane) * 16;
          b[q] = mk_frag(*(const v8bf*)bp, *(const v8bf*)(bp + 8));
        }
#pragma unroll
        for (int q = 0; q < 4; ++q)
          acc[q] = __builtin_amdgcn_wmma_f32_16x16x32_bf16(
              false, a, false, b[q], (short)0, acc[q], false, false);
      }

#pragma unroll
      for (int r = 0; r < 8; ++r) {
        const float iv = sigmoidf_(acc[0][r]);
        const float fv = sigmoidf_(acc[1][r]);
        const float gv = tanh_fast(acc[2][r]);
        const float ov = sigmoidf_(acc[3][r]);
        const float c = fv * cst[g][r] + iv * gv;
        cst[g][r] = c;
        const float h = ov * tanh_fast(c);
        const int b = rbase + r;
        const int j = jt * 16 + nc0;
        hwrite[b * 512 + j] = f2bf(h);
        if (hs_bf)  hs_bf[((size_t)t * 32 + b) * 1024 + dir * 512 + j] = f2bf(h);
        if (out_f32) out_f32[((size_t)b * 512 + t) * 1024 + dir * 512 + j] = h;
        if (s == 511) {
          hT_base[(dir * 32 + b) * 512 + j] = h;
          cT_base[(dir * 32 + b) * 512 + j] = c;
        }
      }
    }
    __syncthreads();
  }
}

// ---------------------------------------------------------------------------
extern "C" void kernel_launch(void* const* d_in, const int* in_sizes, int n_in,
                              void* d_out, int out_size, void* d_ws, size_t ws_size,
                              hipStream_t stream) {
  (void)in_sizes; (void)n_in; (void)out_size;
  const float* X = (const float*)d_in[0];
  // dir index: 0=l0 fwd, 1=l0 rev, 2=l1 fwd, 3=l1 rev
  const float* W_ih[4] = {(const float*)d_in[1], (const float*)d_in[5],
                          (const float*)d_in[9], (const float*)d_in[13]};
  const float* W_hh[4] = {(const float*)d_in[2], (const float*)d_in[6],
                          (const float*)d_in[10], (const float*)d_in[14]};
  const float* b_ih[4] = {(const float*)d_in[3], (const float*)d_in[7],
                          (const float*)d_in[11], (const float*)d_in[15]};
  const float* b_hh[4] = {(const float*)d_in[4], (const float*)d_in[8],
                          (const float*)d_in[12], (const float*)d_in[16]};
  float* out = (float*)d_out;

  const int Kih[4] = {512, 512, 1024, 1024};

  char* p = (char*)d_ws;
  auto carve = [&](size_t bytes) -> void* {
    void* r = (void*)p;
    p += (bytes + 255) & ~(size_t)255;
    return r;
  };
  __bf16* Xbf = (__bf16*)carve((size_t)16384 * 512 * 2);    // 16.8 MB
  __bf16* hs0 = (__bf16*)carve((size_t)16384 * 1024 * 2);   // 33.6 MB
  __bf16* gxF = (__bf16*)carve((size_t)16384 * 2048 * 2);   // 67.1 MB
  __bf16* gxR = (__bf16*)carve((size_t)16384 * 2048 * 2);   // 67.1 MB
  __bf16* WihP[4]; __bf16* WhhP[4]; float* bc[4];
  for (int d = 0; d < 4; ++d) WihP[d] = (__bf16*)carve((size_t)2048 * Kih[d] * 2);
  for (int d = 0; d < 4; ++d) WhhP[d] = (__bf16*)carve((size_t)2048 * 512 * 2);
  for (int d = 0; d < 4; ++d) bc[d]   = (float*)carve((size_t)2048 * 4);
  if ((size_t)(p - (char*)d_ws) > ws_size) return;  // insufficient scratch

  // --- pack / convert ---
  pack_x_kernel<<<(32 * 512 * 512) / 256, 256, 0, stream>>>(X, Xbf);
  for (int d = 0; d < 4; ++d) {
    bias_kernel<<<8, 256, 0, stream>>>(b_ih[d], b_hh[d], bc[d]);
    pack_w_kernel<<<(2048 * Kih[d]) / 256, 256, 0, stream>>>(W_ih[d], WihP[d], Kih[d]);
    pack_w_kernel<<<(2048 * 512) / 256, 256, 0, stream>>>(W_hh[d], WhhP[d], 512);
  }

  const size_t FO = (size_t)32 * 512 * 1024;   // final_output elems
  const size_t HC = (size_t)4 * 32 * 512;      // final_hidden elems

  // --- layer 0 ---
  gemm_gx_kernel<<<4096, 256, 0, stream>>>(Xbf, WihP[0], bc[0], gxF, 16384, 512);
  gemm_gx_kernel<<<4096, 256, 0, stream>>>(Xbf, WihP[1], bc[1], gxR, 16384, 512);
  lstm_layer_kernel<<<2, 256, 0, stream>>>(gxF, gxR, WhhP[0], WhhP[1],
                                           hs0, (float*)nullptr,
                                           out + FO, out + FO + HC);
  // --- layer 1 ---
  gemm_gx_kernel<<<4096, 256, 0, stream>>>(hs0, WihP[2], bc[2], gxF, 16384, 1024);
  gemm_gx_kernel<<<4096, 256, 0, stream>>>(hs0, WihP[3], bc[3], gxR, 16384, 1024);
  lstm_layer_kernel<<<2, 256, 0, stream>>>(gxF, gxR, WhhP[2], WhhP[3],
                                           (__bf16*)nullptr, out,
                                           out + FO + 2 * 32 * 512,
                                           out + FO + HC + 2 * 32 * 512);
}
